// SAGE_34634616275240
// MI455X (gfx1250) — matrix operand order repaired and checked by reading, not compiled, not run on hardware
//
#include <hip/hip_runtime.h>

#define NN 50000
#define NE 800000
#define IC 128
#define HD 16
#define OC 64
#define NTILES ((NN + 15) / 16)   // 3125
#define EPW 4                     // edges per wave in agg1

typedef __attribute__((ext_vector_type(16))) _Float16 v16h;
typedef __attribute__((ext_vector_type(8)))  float    v8f;

// ---------------- workspace layout (bytes, 32B-aligned sections) ----------
// msg1 : NN*IC*4   = 25,600,000   @ 0
// deg  : NN*4      =    200,000   @ 25,600,000  (padded to 200,192)
// h    : NN*HD*4   =  3,200,000   @ 25,800,192
// msg2 : NN*HD*4   =  3,200,000   @ 29,000,192
// bw1  : 4096*2    =      8,192   @ 32,200,192  (pre-swizzled [W1l;W1r] f16)
// bw2  : 2048*2    =      4,096   @ 32,208,384  (pre-swizzled [W2l;W2r] f16)
// total            = 32,212,480
#define OFF_DEG  25600000
#define OFF_H    25800192
#define OFF_MSG2 29000192
#define OFF_BW1  32200192
#define OFF_BW2  32208384
#define WS_ZERO_FLOATS (32200192 / 4)   // zero accumulators only (not bw)

// -------------------------------------------------------------------------
__global__ void __launch_bounds__(256) zero_ws_kernel(float* __restrict__ ws, int n) {
  int i = blockIdx.x * blockDim.x + threadIdx.x;
  int stride = gridDim.x * blockDim.x;
  for (; i < n; i += stride) ws[i] = 0.0f;
}

// Pre-swizzle weights into the exact per-lane WMMA B layout:
//   bw1[ch*512 + lane*16 + j] = Wcat1[32*ch + 16*(j>>3) + 8*(lane>>4) + (j&7)][lane&15]
//   bw2[ t*512 + lane*16 + j] = Wcat2[16*(j>>3) + 8*(lane>>4) + (j&7)][16*t + (lane&15)]
// so the hot kernels read B as one 32-byte contiguous v16h per lane.
__global__ void __launch_bounds__(256) swizzle_weights_kernel(
    const float* __restrict__ W1l, const float* __restrict__ W1r,
    const float* __restrict__ W2l, const float* __restrict__ W2r,
    _Float16* __restrict__ bw1, _Float16* __restrict__ bw2) {
  int i = blockIdx.x * blockDim.x + threadIdx.x;   // 6144 threads exactly
  if (i < 4096) {
    int ch = i >> 9, r = i & 511, lane = r >> 4, j = r & 15;
    int m = lane & 15, g = lane >> 4;
    int k = 32 * ch + 16 * (j >> 3) + 8 * g + (j & 7); // combined K in [0,256)
    float v = (k < IC) ? W1l[k * HD + m] : W1r[(k - IC) * HD + m];
    bw1[i] = (_Float16)v;
  } else {
    int ii = i - 4096;
    int t = ii >> 9, r = ii & 511, lane = r >> 4, j = r & 15;
    int m = lane & 15, g = lane >> 4;
    const float* W = (j >> 3) ? W2r : W2l;         // K = 16*(j>>3) + 8g + (j&7)
    bw2[ii] = (_Float16)W[(8 * g + (j & 7)) * OC + 16 * t + m];
  }
}

// Layer-1 mean aggregation: one wave per EPW edges, 32 lanes x float4 = 128 ch.
// Prefetch the gather rows (global_prefetch_b8) before the atomic loop.
__global__ void __launch_bounds__(256) agg1_kernel(
    const float* __restrict__ x, const long long* __restrict__ ei,
    float* __restrict__ msg1, float* __restrict__ deg) {
  int gid  = blockIdx.x * blockDim.x + threadIdx.x;
  int wave = gid >> 5;                    // grid sized exactly: (NE/EPW)*32 threads
  int lane = gid & 31;
  int e0   = wave * EPW;
  int src[EPW], dst[EPW];
#pragma unroll
  for (int i = 0; i < EPW; ++i) {
    src[i] = (int)ei[e0 + i];
    dst[i] = (int)ei[NE + e0 + i];
  }
#pragma unroll
  for (int i = 0; i < EPW; ++i)           // per-lane prefetch covers whole row
    __builtin_prefetch(x + (size_t)src[i] * IC + lane * 4, 0, 0);
#pragma unroll
  for (int i = 0; i < EPW; ++i) {
    float4 v = ((const float4*)(x + (size_t)src[i] * IC))[lane];
    float* md = msg1 + (size_t)dst[i] * IC + lane * 4;
    atomicAdd(md + 0, v.x);
    atomicAdd(md + 1, v.y);
    atomicAdd(md + 2, v.z);
    atomicAdd(md + 3, v.w);
    if (lane == 0) atomicAdd(deg + dst[i], 1.0f);
  }
}

// One thread per (edge, channel) for the 16-wide layer-2 aggregation.
__global__ void __launch_bounds__(256) agg2_kernel(
    const float* __restrict__ h, const long long* __restrict__ ei,
    float* __restrict__ msg2) {
  int gid = blockIdx.x * blockDim.x + threadIdx.x;   // NE*16 threads exactly
  int e   = gid >> 4;
  int ch  = gid & 15;
  int src = (int)ei[e];
  int dst = (int)ei[NE + e];
  atomicAdd(msg2 + (size_t)dst * HD + ch, h[(size_t)src * HD + ch]);
}

// Layer 1: h = relu([mean | x] @ [W1l ; W1r] + b1l).  One wave per 16-node
// tile, concatenated K = 256 -> 8 x v_wmma_f32_16x16x32_f16.
__global__ void __launch_bounds__(256) sage_layer1_wmma(
    const float* __restrict__ x,   const _Float16* __restrict__ bw1,
    const float* __restrict__ b1l,
    const float* __restrict__ msg1, const float* __restrict__ deg,
    float* __restrict__ h) {
  int wid = (blockIdx.x * blockDim.x + threadIdx.x) >> 5;
  if (wid >= NTILES) return;                     // wave-uniform: EXEC stays all-1
  int lane = threadIdx.x & 31;
  int m = lane & 15;                             // A row / B,C column
  int g = lane >> 4;                             // lane-group select
  int node = wid * 16 + m;
  int row  = node < NN ? node : NN - 1;          // clamp loads, mask stores
  float rdeg = 1.0f / fmaxf(deg[row], 1.0f);
  const float* mrow = msg1 + (size_t)row * IC;
  const float* xrow = x    + (size_t)row * IC;

  v8f c = {0.f, 0.f, 0.f, 0.f, 0.f, 0.f, 0.f, 0.f};
#pragma unroll
  for (int ch = 0; ch < 8; ++ch) {
    const float* base = (ch < 4) ? mrow : xrow;  // first 128 K = mean, rest = x
    float scale = (ch < 4) ? rdeg : 1.0f;
    int k0 = (ch & 3) * 32;
    // A layout (16-bit, 16x32): elems 0-7 -> K = 8g+j ; elems 8-15 -> K = 16+8g+j
    float4 f0 = *(const float4*)(base + k0 + 8 * g);
    float4 f1 = *(const float4*)(base + k0 + 8 * g + 4);
    float4 f2 = *(const float4*)(base + k0 + 16 + 8 * g);
    float4 f3 = *(const float4*)(base + k0 + 16 + 8 * g + 4);
    v16h a;
    a[0]=(_Float16)(f0.x*scale); a[1]=(_Float16)(f0.y*scale);
    a[2]=(_Float16)(f0.z*scale); a[3]=(_Float16)(f0.w*scale);
    a[4]=(_Float16)(f1.x*scale); a[5]=(_Float16)(f1.y*scale);
    a[6]=(_Float16)(f1.z*scale); a[7]=(_Float16)(f1.w*scale);
    a[8]=(_Float16)(f2.x*scale); a[9]=(_Float16)(f2.y*scale);
    a[10]=(_Float16)(f2.z*scale); a[11]=(_Float16)(f2.w*scale);
    a[12]=(_Float16)(f3.x*scale); a[13]=(_Float16)(f3.y*scale);
    a[14]=(_Float16)(f3.z*scale); a[15]=(_Float16)(f3.w*scale);
    // B: pre-swizzled, one 32-byte contiguous load per lane
    v16h b = *(const v16h*)(bw1 + (ch << 9) + (lane << 4));
    c = __builtin_amdgcn_wmma_f32_16x16x32_f16(false, a, false, b,
                                               (short)0, c, false, false);
  }
  // D layout: elem r -> out row = 16*wid + r + 8g, col = m
  float bias = b1l[m];
#pragma unroll
  for (int r = 0; r < 8; ++r) {
    int orow = wid * 16 + r + 8 * g;
    if (orow < NN) h[(size_t)orow * HD + m] = fmaxf(c[r] + bias, 0.0f);
  }
}

// Layer 2: out = [mean_h | h] @ [W2l ; W2r] + b2l.  Concatenated K = 32 ->
// exactly one WMMA per 16x16 output tile; 4 tiles for OC = 64.
__global__ void __launch_bounds__(256) sage_layer2_wmma(
    const float* __restrict__ h,   const _Float16* __restrict__ bw2,
    const float* __restrict__ b2l,
    const float* __restrict__ msg2, const float* __restrict__ deg,
    float* __restrict__ out) {
  int wid = (blockIdx.x * blockDim.x + threadIdx.x) >> 5;
  if (wid >= NTILES) return;
  int lane = threadIdx.x & 31;
  int m = lane & 15;
  int g = lane >> 4;
  int node = wid * 16 + m;
  int row  = node < NN ? node : NN - 1;
  float rdeg = 1.0f / fmaxf(deg[row], 1.0f);
  // A: elems 0-7 -> K = 8g+j  (mean_h, K<16);  elems 8-15 -> K = 16+8g+j (h)
  const float* m2 = msg2 + (size_t)row * HD + 8 * g;
  const float* hr = h    + (size_t)row * HD + 8 * g;
  float4 f0 = *(const float4*)(m2);
  float4 f1 = *(const float4*)(m2 + 4);
  float4 f2 = *(const float4*)(hr);
  float4 f3 = *(const float4*)(hr + 4);
  v16h a;
  a[0]=(_Float16)(f0.x*rdeg); a[1]=(_Float16)(f0.y*rdeg);
  a[2]=(_Float16)(f0.z*rdeg); a[3]=(_Float16)(f0.w*rdeg);
  a[4]=(_Float16)(f1.x*rdeg); a[5]=(_Float16)(f1.y*rdeg);
  a[6]=(_Float16)(f1.z*rdeg); a[7]=(_Float16)(f1.w*rdeg);
  a[8]=(_Float16)f2.x;  a[9]=(_Float16)f2.y;
  a[10]=(_Float16)f2.z; a[11]=(_Float16)f2.w;
  a[12]=(_Float16)f3.x; a[13]=(_Float16)f3.y;
  a[14]=(_Float16)f3.z; a[15]=(_Float16)f3.w;

  v8f acc[4];
#pragma unroll
  for (int t = 0; t < 4; ++t) {
    v16h b = *(const v16h*)(bw2 + (t << 9) + (lane << 4));
    v8f cz = {0.f, 0.f, 0.f, 0.f, 0.f, 0.f, 0.f, 0.f};
    acc[t] = __builtin_amdgcn_wmma_f32_16x16x32_f16(false, a, false, b,
                                                    (short)0, cz, false, false);
  }
#pragma unroll
  for (int t = 0; t < 4; ++t) {
    float bias = b2l[16 * t + m];
#pragma unroll
    for (int r = 0; r < 8; ++r) {
      int orow = wid * 16 + r + 8 * g;
      if (orow < NN) out[(size_t)orow * OC + 16 * t + m] = acc[t][r] + bias;
    }
  }
}

// Row-wise log_softmax over 64 logits: one wave per row, 2 elems/lane,
// wave32 shuffle reductions. In-place on d_out.
__global__ void __launch_bounds__(256) logsoftmax_kernel(float* __restrict__ out) {
  int gid  = blockIdx.x * blockDim.x + threadIdx.x;  // NN*32 threads exactly
  int rowi = gid >> 5;
  int lane = gid & 31;
  float* p = out + (size_t)rowi * OC;
  float v0 = p[lane];
  float v1 = p[lane + 32];
  float mx = fmaxf(v0, v1);
#pragma unroll
  for (int off = 16; off > 0; off >>= 1)
    mx = fmaxf(mx, __shfl_xor(mx, off, 32));
  float s = __expf(v0 - mx) + __expf(v1 - mx);
#pragma unroll
  for (int off = 16; off > 0; off >>= 1)
    s += __shfl_xor(s, off, 32);
  float lse = mx + __logf(s);
  p[lane]      = v0 - lse;
  p[lane + 32] = v1 - lse;
}

// -------------------------------------------------------------------------
extern "C" void kernel_launch(void* const* d_in, const int* in_sizes, int n_in,
                              void* d_out, int out_size, void* d_ws, size_t ws_size,
                              hipStream_t stream) {
  const float*     x   = (const float*)d_in[0];
  const float*     W1l = (const float*)d_in[1];
  const float*     b1l = (const float*)d_in[2];
  const float*     W1r = (const float*)d_in[3];
  const float*     W2l = (const float*)d_in[4];
  const float*     b2l = (const float*)d_in[5];
  const float*     W2r = (const float*)d_in[6];
  const long long* ei  = (const long long*)d_in[7];   // int64 edge_index [2, NE]
  float* out = (float*)d_out;

  char*      ws   = (char*)d_ws;
  float*     msg1 = (float*)(ws);
  float*     deg  = (float*)(ws + OFF_DEG);
  float*     hbuf = (float*)(ws + OFF_H);
  float*     msg2 = (float*)(ws + OFF_MSG2);
  _Float16*  bw1  = (_Float16*)(ws + OFF_BW1);
  _Float16*  bw2  = (_Float16*)(ws + OFF_BW2);

  // 1) zero accumulators (deterministic every call)
  zero_ws_kernel<<<2048, 256, 0, stream>>>((float*)ws, WS_ZERO_FLOATS);
  // 1b) pre-swizzle weights into WMMA B layout (6144 threads = 24 blocks)
  swizzle_weights_kernel<<<24, 256, 0, stream>>>(W1l, W1r, W2l, W2r, bw1, bw2);
  // 2) layer-1 mean aggregation: wave per EPW edges ((NE/EPW)*32/256 = 25000)
  agg1_kernel<<<(NE / EPW) * 32 / 256, 256, 0, stream>>>(x, ei, msg1, deg);
  // 3) layer-1 fused dual GEMM + bias + relu (WMMA)
  sage_layer1_wmma<<<(NTILES + 7) / 8, 256, 0, stream>>>(x, bw1, b1l,
                                                         msg1, deg, hbuf);
  // 4) layer-2 aggregation: thread per (edge, channel) (NE*16/256 = 50000)
  agg2_kernel<<<(NE * 16) / 256, 256, 0, stream>>>(hbuf, ei, msg2);
  // 5) layer-2 fused dual GEMM + bias (WMMA) -> raw logits in d_out
  sage_layer2_wmma<<<(NTILES + 7) / 8, 256, 0, stream>>>(hbuf, bw2, b2l,
                                                         msg2, deg, out);
  // 6) in-place log_softmax (NN*32/256 = 6250 blocks exact)
  logsoftmax_kernel<<<(NN * 32) / 256, 256, 0, stream>>>(out);
}
